// C2Former_22041772163308
// MI455X (gfx1250) — compile-verified, matrix operand order checked by reading, not run on hardware
//
#include <hip/hip_runtime.h>
#include <math.h>

// ---------------------------------------------------------------------------
// CDNA5 (gfx1250) WMMA helpers: bf16 A/B, f32 accumulate, 16x16x32 shape.
// ---------------------------------------------------------------------------
typedef __attribute__((ext_vector_type(16))) __bf16 v16bf;
typedef __attribute__((ext_vector_type(8)))  float  v8f;
typedef __attribute__((ext_vector_type(4)))  unsigned v4u;
typedef __attribute__((ext_vector_type(8)))  int      v8i;
typedef __attribute__((ext_vector_type(4)))  int      v4i;

union Frag16 { v16bf v; unsigned u[8]; };

__device__ __forceinline__ unsigned f32_to_bf16u(float f) {
  unsigned u = __float_as_uint(f);
  return (u + 0x7FFFu + ((u >> 16) & 1u)) >> 16;   // round-to-nearest-even
}
__device__ __forceinline__ unsigned pack_bf16x2(float lo, float hi) {
  return f32_to_bf16u(lo) | (f32_to_bf16u(hi) << 16);
}
// K index of bf16 element-pair j (0..7) for a lane-half (A 16x32 / B 32x16 layout)
__device__ __forceinline__ int kbase(int j, int half) {
  return (j < 4 ? 2 * j : 16 + 2 * (j - 4)) + 8 * half;
}

// ---------------------------------------------------------------------------
// Tensor Data Mover: 2-D tile (tile_x cols x tile_y rows of f32) from global
// [row_stride f32 pitch] into LDS with +1 DWORD row padding (pad_interval /
// pad_amount), OOB rows/cols zero-filled by hardware (tensor_dim = remaining
// extent). Issued by one wave; completion via TENSORcnt.
// D# layout per CDNA5 ISA ch.8 (group0 128b, group1 256b, groups2/3 zero=2D).
// This toolchain's builtin takes 6 args: (v4u g0, v8i g1, v4i g2, v4i g3,
// v8i extra, i32 cpol) -- extra group zeroed for 2-D tiles.
// ---------------------------------------------------------------------------
#if __has_builtin(__builtin_amdgcn_tensor_load_to_lds) && \
    __has_builtin(__builtin_amdgcn_s_wait_tensorcnt)
#define HAVE_TDM 1
#endif

__device__ __forceinline__ void tdm_load_2d(float* ldsp, const float* gptr,
                                            unsigned tile_x, unsigned tile_y,
                                            unsigned ext_x, unsigned ext_y,
                                            unsigned row_stride, unsigned lds_stride,
                                            unsigned pad_code, int lane)
{
#ifdef HAVE_TDM
  (void)lds_stride; (void)lane;
  unsigned long long ga = (unsigned long long)(uintptr_t)gptr;
  unsigned lds_addr = (unsigned)(uintptr_t)ldsp;      // low 32 bits = LDS byte offset
  v4u g0;
  g0[0] = 1u;                                         // count=1 (user descriptor)
  g0[1] = lds_addr;                                   // [63:32] lds_addr
  g0[2] = (unsigned)(ga & 0xFFFFFFFFull);             // [95:64] global_addr lo
  g0[3] = (unsigned)((ga >> 32) & 0x01FFFFFFull)      // [120:96] global_addr hi
        | (2u << 30);                                 // [127:126] type=2 (image)
  v8i g1;
  g1[0] = (int)((2u << 16)                            // data_size = 4B
              | (1u << 20)                            // pad_enable
              | (pad_code << 22));                    // pad_interval (tile row), amount=1 DW
  g1[1] = (int)((ext_x & 0xFFFFu) << 16);             // tensor_dim0 [15:0]  (abar addr = 0)
  g1[2] = (int)((ext_x >> 16) | ((ext_y & 0xFFFFu) << 16));   // dim0 hi | dim1 lo
  g1[3] = (int)((ext_y >> 16) | (tile_x << 16));      // dim1 hi | tile_dim0
  g1[4] = (int)(tile_y & 0xFFFFu);                    // tile_dim1 (tile_dim2 = 0)
  g1[5] = (int)row_stride;                            // tensor_dim0_stride lo32
  g1[6] = 0;                                          // stride hi16 | dim1_stride lo16
  g1[7] = 0;
  v4i gz;  gz[0] = 0; gz[1] = 0; gz[2] = 0; gz[3] = 0;   // groups 2/3 unused (2-D)
  v8i gz8; for (int q = 0; q < 8; q++) gz8[q] = 0;
  __builtin_amdgcn_tensor_load_to_lds(g0, g1, gz, gz, gz8, 0);
#else
  (void)pad_code;
  for (unsigned idx = (unsigned)lane; idx < tile_x * tile_y; idx += 32) {
    unsigned y = idx / tile_x, x = idx - y * tile_x;
    float v = (x < ext_x && y < ext_y) ? gptr[(long)y * row_stride + x] : 0.f;
    ldsp[y * lds_stride + x] = v;
  }
#endif
}

__device__ __forceinline__ void tdm_wait() {
#ifdef HAVE_TDM
  __builtin_amdgcn_s_wait_tensorcnt(0);
#endif
}

// ---------------------------------------------------------------------------
// Generic WMMA GEMM:  Y[b] = epi(W @ X[b] + bias)  with M mult 64, K mult 32.
// BMODE: 0 = dense X [K][Nb], 1 = concat(X1,X2) along K, 2 = im2col 3x3 pad 1
// EPI:   0 = none, 1 = relu, 2 = relu + residual add
// Block: 128 threads = 4 waves (2x2), block tile 64x64, K step 32.
// A (weights) and dense B tiles staged by TDM as f32 (padded rows); bf16
// conversion fused into the fragment loads.
// ---------------------------------------------------------------------------
template<int BMODE, int EPI>
__global__ void __launch_bounds__(128) gemm_wmma(
    const float* __restrict__ X1, const float* __restrict__ X2,
    const float* __restrict__ Wt, const float* __restrict__ bias,
    float* __restrict__ Y, const float* __restrict__ Res,
    int M, int K, int K1, int Nb, long xbs, long ybs, long rbs,
    int Hs, int Wsp)
{
  __shared__ float Af[64 * 33];   // A tile: 64 rows x 32 f32, row stride 33
  __shared__ float Bf[32 * 65];   // B tile: 32 k-rows x 64 n f32, row stride 65

  const int tid  = threadIdx.x;
  const int wave = tid >> 5, lane = tid & 31, half = lane >> 4, ln = lane & 15;
  const int wm = wave >> 1, wn = wave & 1;
  const int m0 = blockIdx.y * 64, n0 = blockIdx.x * 64, bz = blockIdx.z;

  const float* X1b = X1 + (long)bz * xbs;
  const float* X2b = (BMODE == 1) ? (X2 + (long)bz * xbs) : nullptr;

  v8f acc[2][2];
  for (int i = 0; i < 2; i++)
    for (int j = 0; j < 2; j++)
      for (int r = 0; r < 8; r++) acc[i][j][r] = 0.f;

  for (int k0 = 0; k0 < K; k0 += 32) {
    // A tile (weights, [M][K] row-major) via TDM, issued by wave 0
    if (wave == 0)
      tdm_load_2d(Af, Wt + (long)m0 * K + k0, 32, 64,
                  (unsigned)(K - k0), (unsigned)(M - m0), (unsigned)K, 33, 4, lane);
    if (BMODE == 2) {
      // im2col gather fill of B (overlaps the A-tile DMA); one k-row / thread
      int kk = k0 + (tid >> 2);
      int c = kk / 9, t = kk - c * 9;
      int dy = t / 3 - 1, dx = (t % 3) - 1;
      const float* Xc = X1b + (long)c * Hs * Wsp;
      int bn = n0 + (tid & 3) * 16;
      float* brow = &Bf[(tid >> 2) * 65 + (tid & 3) * 16];
      #pragma unroll
      for (int e = 0; e < 16; e++) {
        int n = bn + e;
        int y = n / Wsp + dy, x = n - (n / Wsp) * Wsp + dx;
        float v = 0.f;
        if (n < Nb && (unsigned)y < (unsigned)Hs && (unsigned)x < (unsigned)Wsp)
          v = Xc[y * Wsp + x];
        brow[e] = v;
      }
    } else if (wave == 0) {
      // dense / concat B tile via TDM (K1 multiple of 32 -> no straddling)
      const float* src = (BMODE == 1 && k0 >= K1)
                           ? X2b + (long)(k0 - K1) * Nb + n0
                           : X1b + (long)k0 * Nb + n0;
      tdm_load_2d(Bf, src, 64, 32,
                  (unsigned)(Nb - n0), (unsigned)(K - k0), (unsigned)Nb, 65, 5, lane);
    }
    if (wave == 0) tdm_wait();
    __syncthreads();

    Frag16 af[2], bfr[2];
    #pragma unroll
    for (int i = 0; i < 2; i++) {
      int ar = wm * 32 + i * 16 + ln;
      const float* arow = &Af[ar * 33];
      #pragma unroll
      for (int j = 0; j < 8; j++) {
        int kb = kbase(j, half);
        af[i].u[j] = pack_bf16x2(arow[kb], arow[kb + 1]);
      }
    }
    #pragma unroll
    for (int jf = 0; jf < 2; jf++) {
      int bn = wn * 32 + jf * 16 + ln;
      #pragma unroll
      for (int j = 0; j < 8; j++) {
        int kb = kbase(j, half);
        bfr[jf].u[j] = pack_bf16x2(Bf[kb * 65 + bn], Bf[(kb + 1) * 65 + bn]);
      }
    }
    #pragma unroll
    for (int i = 0; i < 2; i++)
      #pragma unroll
      for (int j = 0; j < 2; j++)
        acc[i][j] = __builtin_amdgcn_wmma_f32_16x16x32_bf16(
            false, af[i].v, false, bfr[j].v, (short)0, acc[i][j], false, false);
    __syncthreads();
  }

  float* Yb = Y + (long)bz * ybs;
  const float* Rb = (EPI == 2) ? (Res + (long)bz * rbs) : nullptr;
  #pragma unroll
  for (int i = 0; i < 2; i++) {
    #pragma unroll
    for (int j = 0; j < 2; j++) {
      int n = n0 + wn * 32 + j * 16 + ln;
      if (n >= Nb) continue;
      #pragma unroll
      for (int r = 0; r < 8; r++) {
        int m = m0 + wm * 32 + i * 16 + r + 8 * half;
        float v = acc[i][j][r];
        if (bias) v += bias[m];
        if (EPI >= 1) v = fmaxf(v, 0.f);
        if (EPI == 2) v += Rb[(long)m * Nb + n];
        Yb[(long)m * Nb + n] = v;
      }
    }
  }
}

// ---------------------------------------------------------------------------
// Fused attention: per wave, 16 query rows; S(16x864) via 54 WMMAs into LDS,
// exact f32 softmax (1/rowsum deferred to epilogue), then P@V^T via 27x2 WMMAs.
// grid (240, 12), block 64 (2 waves), dyn LDS = (2*16*872 + 32) floats.
// ---------------------------------------------------------------------------
__global__ void __launch_bounds__(64) attn_kernel(
    const float* __restrict__ Q, const float* __restrict__ Km,
    const float* __restrict__ Vm, float* __restrict__ O)
{
  const int HW = 7680, NS = 864, C = 192;
  extern __shared__ float smem[];
  const int head = blockIdx.y;
  const int b = head / 6, cbase = (head % 6) * 32;
  const int wave = threadIdx.x >> 5, lane = threadIdx.x & 31;
  const int half = lane >> 4, ln = lane & 15;
  float* S  = smem + wave * (16 * 872);
  float* rs = smem + 2 * (16 * 872) + wave * 16;
  const int m0 = (blockIdx.x * 2 + wave) * 16;

  const float* qp = Q  + ((long)b * C + cbase) * HW;
  const float* kp = Km + ((long)b * C + cbase) * NS;
  const float* vp = Vm + ((long)b * C + cbase) * NS;

  Frag16 aq;                 // A = Q^T tile [16 m][32 c]
  {
    int m = m0 + ln;
    #pragma unroll
    for (int j = 0; j < 8; j++) {
      int c = kbase(j, half);
      aq.u[j] = pack_bf16x2(qp[(long)c * HW + m], qp[(long)(c + 1) * HW + m]);
    }
  }
  const float scale = 0.1767766952966369f;   // 32^-0.5
  for (int nt = 0; nt < 54; nt++) {
    Frag16 bk;               // B = K [32 c][16 n]
    int n = nt * 16 + ln;
    #pragma unroll
    for (int j = 0; j < 8; j++) {
      int c = kbase(j, half);
      bk.u[j] = pack_bf16x2(kp[(long)c * NS + n], kp[(long)(c + 1) * NS + n]);
    }
    v8f s;
    for (int r = 0; r < 8; r++) s[r] = 0.f;
    s = __builtin_amdgcn_wmma_f32_16x16x32_bf16(false, aq.v, false, bk.v,
                                                (short)0, s, false, false);
    #pragma unroll
    for (int r = 0; r < 8; r++) S[(r + 8 * half) * 872 + n] = s[r] * scale;
  }
  __syncthreads();
  { // softmax: 2 lanes per row, 432 cols each; combine halves via shfl_xor(16)
    int r = ln, c0 = half * 432;
    float mx = -3.0e38f;
    for (int c = 0; c < 432; c++) mx = fmaxf(mx, S[r * 872 + c0 + c]);
    mx = fmaxf(mx, __shfl_xor(mx, 16, 32));
    float sum = 0.f;
    for (int c = 0; c < 432; c++) {
      float e = __expf(S[r * 872 + c0 + c] - mx);
      S[r * 872 + c0 + c] = e; sum += e;
    }
    sum += __shfl_xor(sum, 16, 32);
    if (half == 0) rs[r] = sum;
  }
  __syncthreads();

  v8f o0, o1;                // O tile 16(m) x 32(ch)
  for (int r = 0; r < 8; r++) { o0[r] = 0.f; o1[r] = 0.f; }
  for (int ks = 0; ks < 27; ks++) {
    Frag16 ap, bv0, bv1;
    #pragma unroll
    for (int j = 0; j < 8; j++) {
      int n = ks * 32 + kbase(j, half);
      ap.u[j]  = pack_bf16x2(S[ln * 872 + n], S[ln * 872 + n + 1]);
      bv0.u[j] = pack_bf16x2(vp[(long)ln * NS + n],        vp[(long)ln * NS + n + 1]);
      bv1.u[j] = pack_bf16x2(vp[(long)(ln + 16) * NS + n], vp[(long)(ln + 16) * NS + n + 1]);
    }
    o0 = __builtin_amdgcn_wmma_f32_16x16x32_bf16(false, ap.v, false, bv0.v,
                                                 (short)0, o0, false, false);
    o1 = __builtin_amdgcn_wmma_f32_16x16x32_bf16(false, ap.v, false, bv1.v,
                                                 (short)0, o1, false, false);
  }
  float* op = O + ((long)b * C + cbase) * HW;
  #pragma unroll
  for (int r = 0; r < 8; r++) {
    int row = r + 8 * half;
    float inv = 1.f / rs[row];
    op[(long)ln * HW        + m0 + row] = o0[r] * inv;
    op[(long)(ln + 16) * HW + m0 + row] = o1[r] * inv;
  }
}

// ---------------------------------------------------------------------------
// Small VALU kernels
// ---------------------------------------------------------------------------
__global__ void k_dwconv(const float* __restrict__ x, const float* __restrict__ w,
                         const float* __restrict__ bias, float* __restrict__ y)
{
  int idx = blockIdx.x * blockDim.x + threadIdx.x;
  if (idx >= 2 * 192 * 864) return;
  int i = idx % 864, c = (idx / 864) % 192, b = idx / (192 * 864);
  int yk = i / 32, xk = i % 32;
  float acc = bias[c];
  const float* xb = x + ((long)b * 192 + c) * 7680;
  const float* wc = w + c * 49;
  for (int ky = 0; ky < 7; ky++) {
    int iy = yk * 3 + ky - 3;
    if ((unsigned)iy >= 80u) continue;
    for (int kx = 0; kx < 7; kx++) {
      int ix = xk * 3 + kx - 3;
      if ((unsigned)ix >= 96u) continue;
      acc += wc[ky * 7 + kx] * xb[iy * 96 + ix];
    }
  }
  y[idx] = acc;
}

__global__ void k_ln_gelu(float* __restrict__ dw, const float* __restrict__ g,
                          const float* __restrict__ bb)
{
  __shared__ float red[256];
  int b = blockIdx.x / 864, pos = blockIdx.x % 864, c = threadIdx.x;
  float v = (c < 192) ? dw[((long)b * 192 + c) * 864 + pos] : 0.f;
  red[c] = v; __syncthreads();
  for (int st = 128; st > 0; st >>= 1) { if (c < st) red[c] += red[c + st]; __syncthreads(); }
  float m = red[0] / 192.f; __syncthreads();
  red[c] = v * v; __syncthreads();
  for (int st = 128; st > 0; st >>= 1) { if (c < st) red[c] += red[c + st]; __syncthreads(); }
  float var = red[0] / 192.f - m * m;
  if (c < 192) {
    float xn = (v - m) * rsqrtf(var + 1e-5f) * g[c] + bb[c];
    float ge = 0.5f * xn * (1.f + erff(xn * 0.70710678118654752f));  // exact GELU
    dw[((long)b * 192 + c) * 864 + pos] = ge;
  }
}

__global__ void k_offsets(const float* __restrict__ x, const float* __restrict__ pw,
                          float* __restrict__ off)
{
  int idx = blockIdx.x * blockDim.x + threadIdx.x;
  if (idx >= 2 * 2 * 864) return;
  int pos = idx % 864, o = (idx / 864) % 2, b = idx / (2 * 864);
  float acc = 0.f;
  const float* xb = x + (long)b * 192 * 864 + pos;
  for (int c = 0; c < 192; c++) acc += pw[o * 192 + c] * xb[(long)c * 864];
  float rng = (o == 0) ? (1.f / 27.f) : (1.f / 32.f);
  off[idx] = tanhf(acc) * rng * 2.f;
}

__global__ void k_sample(const float* __restrict__ img, const float* __restrict__ offs,
                         float* __restrict__ out, int useoff)
{
  int idx = blockIdx.x * blockDim.x + threadIdx.x;
  if (idx >= 2 * 192 * 864) return;
  int i = idx % 864, c = (idx / 864) % 192, b = idx / (192 * 864);
  int yk = i / 32, xk = i % 32;
  float py = (yk + 0.5f) / 27.f * 2.f - 1.f;
  float px = (xk + 0.5f) / 32.f * 2.f - 1.f;
  if (useoff) { py += offs[((long)b * 2 + 0) * 864 + i]; px += offs[((long)b * 2 + 1) * 864 + i]; }
  float gx = (px + 1.f) * 0.5f * 95.f;
  float gy = (py + 1.f) * 0.5f * 79.f;
  float x0f = floorf(gx), y0f = floorf(gy);
  float wx = gx - x0f, wy = gy - y0f;
  const float* im = img + ((long)b * 192 + c) * 7680;
  int x0 = (int)x0f, y0 = (int)y0f;
  auto G = [&](int xi, int yi) -> float {
    if (xi < 0 || xi > 95 || yi < 0 || yi > 79) return 0.f;
    return im[yi * 96 + xi];
  };
  out[idx] = G(x0, y0) * (1 - wx) * (1 - wy) + G(x0 + 1, y0) * wx * (1 - wy)
           + G(x0, y0 + 1) * (1 - wx) * wy   + G(x0 + 1, y0 + 1) * wx * wy;
}

__global__ void k_stats(const float* __restrict__ x, float* __restrict__ mean,
                        float* __restrict__ var, int Nb)
{
  __shared__ float ss[256], sq[256];
  int bc = blockIdx.x;
  const float* p = x + (long)bc * Nb;
  float s = 0.f, q = 0.f;
  for (int i = threadIdx.x; i < Nb; i += 256) { float v = p[i]; s += v; q += v * v; }
  ss[threadIdx.x] = s; sq[threadIdx.x] = q; __syncthreads();
  for (int st = 128; st > 0; st >>= 1) {
    if (threadIdx.x < st) { ss[threadIdx.x] += ss[threadIdx.x + st];
                            sq[threadIdx.x] += sq[threadIdx.x + st]; }
    __syncthreads();
  }
  if (threadIdx.x == 0) {
    float m = ss[0] / Nb;
    mean[bc] = m;
    var[bc]  = sq[0] / Nb - m * m;
  }
}

__global__ void k_mn_combine(const float* __restrict__ ref, const float* __restrict__ gam,
                             const float* __restrict__ bet, const float* __restrict__ mean_ref,
                             const float* __restrict__ var_ref, const float* __restrict__ mean_lr,
                             const float* __restrict__ var_lr, float* __restrict__ out)
{
  long idx = (long)blockIdx.x * blockDim.x + threadIdx.x;
  if (idx >= 2l * 192 * 7680) return;
  int bc = (int)(idx / 7680);
  float refn = (ref[idx] - mean_ref[bc]) * rsqrtf(var_ref[bc] + 1e-5f);
  float s = sqrtf(fmaxf(var_lr[bc], 0.f) * (7680.f / 7679.f));   // unbiased std
  out[idx] = refn * (gam[idx] + s) + bet[idx] + mean_lr[bc];
}

// ---------------------------------------------------------------------------
// Launcher: full forward pass, stream-ordered.
// ---------------------------------------------------------------------------
extern "C" void kernel_launch(void* const* d_in, const int* in_sizes, int n_in,
                              void* d_out, int out_size, void* d_ws, size_t ws_size,
                              hipStream_t stream)
{
  (void)in_sizes; (void)n_in; (void)out_size; (void)ws_size;
  const float* vis_x  = (const float*)d_in[0];
  const float* lwir_x = (const float*)d_in[1];
  auto P = [&](int i) { return (const float*)d_in[2 + i]; };
  const float *vin_w = P(0),  *vin_b = P(1),  *lin_w = P(2),  *lin_b = P(3),
              *vout_w = P(4), *vout_b = P(5), *lout_w = P(6), *lout_b = P(7),
              *comb_w = P(8), *comb_b = P(9),
              *off_dw_w = P(10), *off_dw_b = P(11),
              *off_ln_g = P(12), *off_ln_b = P(13), *off_pw_w = P(14),
              *pq_lwir_w = P(15), *pq_lwir_b = P(16), *pq_vis_w = P(17), *pq_vis_b = P(18),
              *pk_vis_w = P(19),  *pk_vis_b = P(20),  *pv_vis_w = P(21), *pv_vis_b = P(22),
              *pk_lwir_w = P(23), *pk_lwir_b = P(24), *pv_lwir_w = P(25), *pv_lwir_b = P(26),
              *pout_vis_w = P(27), *pout_vis_b = P(28), *pout_lwir_w = P(29), *pout_lwir_b = P(30),
              *mnv_c_w = P(31), *mnv_c_b = P(32), *mnv_g_w = P(33), *mnv_g_b = P(34),
              *mnv_b_w = P(35), *mnv_b_b = P(36),
              *mnl_c_w = P(37), *mnl_c_b = P(38), *mnl_g_w = P(39), *mnl_g_b = P(40),
              *mnl_b_w = P(41), *mnl_b_b = P(42);

  float* ws = (float*)d_ws;
  size_t pos = 0;
  auto alloc = [&](size_t n) { float* p = ws + pos; pos += n; return p; };
  const size_t BIG = 2ull * 192 * 7680;
  float* vis    = alloc(BIG);
  float* lwir   = alloc(BIG);
  float* gammaB = alloc(BIG);  float* comb = gammaB;   // comb dead before gamma conv
  float* betaB  = alloc(BIG);
  float* xbuf   = alloc(BIG);
  float* qinA   = alloc(BIG);  // mn_vis result -> later attn_out_vis
  float* qinB   = alloc(BIG);  // mn_lwir result -> later attn_out_lwir
  float* qA     = alloc(BIG);  // q_lwir -> later proj_vis
  float* qB     = alloc(BIG);  // q_vis  -> later proj_lwir
  float* dw     = alloc(2 * 192 * 864);
  float* offb   = alloc(2 * 2 * 864);
  float* vis_s  = alloc(2 * 192 * 864);
  float* lwir_s = alloc(2 * 192 * 864);
  float* kvis   = alloc(2 * 192 * 864);
  float* vvis   = alloc(2 * 192 * 864);
  float* klw    = alloc(2 * 192 * 864);
  float* vlw    = alloc(2 * 192 * 864);
  float* mean_v = alloc(384);  float* var_v = alloc(384);
  float* mean_l = alloc(384);  float* var_l = alloc(384);

  dim3 blk(128);
  #define GEMM(BM, EPI, X1, X2, Wt, Bi, Yp, Rp, M_, K_, K1_, Nb_, xbs_, ybs_, rbs_, Hs_, Ws_) \
    gemm_wmma<BM, EPI><<<dim3(((Nb_) + 63) / 64, (M_) / 64, 2), blk, 0, stream>>>(          \
        X1, X2, Wt, Bi, Yp, Rp, M_, K_, K1_, Nb_, (long)(xbs_), (long)(ybs_), (long)(rbs_), Hs_, Ws_)

  // input projections + relu
  GEMM(0, 1, vis_x,  nullptr, vin_w, vin_b, vis,  nullptr, 192, 512, 0, 7680, 512 * 7680, 192 * 7680, 0, 0, 0);
  GEMM(0, 1, lwir_x, nullptr, lin_w, lin_b, lwir, nullptr, 192, 512, 0, 7680, 512 * 7680, 192 * 7680, 0, 0, 0);
  // comb = 1x1 over concat(vis, lwir)
  GEMM(1, 0, vis, lwir, comb_w, comb_b, comb, nullptr, 192, 384, 192, 7680, 192 * 7680, 192 * 7680, 0, 0, 0);
  // offset branch
  k_dwconv <<<(2 * 192 * 864 + 255) / 256, 256, 0, stream>>>(comb, off_dw_w, off_dw_b, dw);
  k_ln_gelu<<<1728, 256, 0, stream>>>(dw, off_ln_g, off_ln_b);
  k_offsets<<<(2 * 2 * 864 + 255) / 256, 256, 0, stream>>>(dw, off_pw_w, offb);
  // bilinear sampling
  k_sample<<<(2 * 192 * 864 + 255) / 256, 256, 0, stream>>>(vis,  offb, vis_s,  1);
  k_sample<<<(2 * 192 * 864 + 255) / 256, 256, 0, stream>>>(lwir, offb, lwir_s, 0);
  // per-channel stats
  k_stats<<<384, 256, 0, stream>>>(vis,  mean_v, var_v, 7680);
  k_stats<<<384, 256, 0, stream>>>(lwir, mean_l, var_l, 7680);
  // modality_norm(vis, lwir) -> qinA  (3x3 convs as im2col WMMA GEMMs, K=1728)
  GEMM(2, 1, vis,  nullptr, mnv_c_w, mnv_c_b, xbuf,   nullptr, 192, 1728, 0, 7680, 192 * 7680, 192 * 7680, 0, 80, 96);
  GEMM(2, 0, xbuf, nullptr, mnv_g_w, mnv_g_b, gammaB, nullptr, 192, 1728, 0, 7680, 192 * 7680, 192 * 7680, 0, 80, 96);
  GEMM(2, 0, xbuf, nullptr, mnv_b_w, mnv_b_b, betaB,  nullptr, 192, 1728, 0, 7680, 192 * 7680, 192 * 7680, 0, 80, 96);
  k_mn_combine<<<(2 * 192 * 7680 + 255) / 256, 256, 0, stream>>>(
      lwir, gammaB, betaB, mean_l, var_l, mean_v, var_v, qinA);
  // modality_norm(lwir, vis) -> qinB
  GEMM(2, 1, lwir, nullptr, mnl_c_w, mnl_c_b, xbuf,   nullptr, 192, 1728, 0, 7680, 192 * 7680, 192 * 7680, 0, 80, 96);
  GEMM(2, 0, xbuf, nullptr, mnl_g_w, mnl_g_b, gammaB, nullptr, 192, 1728, 0, 7680, 192 * 7680, 192 * 7680, 0, 80, 96);
  GEMM(2, 0, xbuf, nullptr, mnl_b_w, mnl_b_b, betaB,  nullptr, 192, 1728, 0, 7680, 192 * 7680, 192 * 7680, 0, 80, 96);
  k_mn_combine<<<(2 * 192 * 7680 + 255) / 256, 256, 0, stream>>>(
      vis, gammaB, betaB, mean_v, var_v, mean_l, var_l, qinB);
  // q/k/v projections
  GEMM(0, 0, qinA,   nullptr, pq_lwir_w, pq_lwir_b, qA,   nullptr, 192, 192, 0, 7680, 192 * 7680, 192 * 7680, 0, 0, 0);
  GEMM(0, 0, qinB,   nullptr, pq_vis_w,  pq_vis_b,  qB,   nullptr, 192, 192, 0, 7680, 192 * 7680, 192 * 7680, 0, 0, 0);
  GEMM(0, 0, vis_s,  nullptr, pk_vis_w,  pk_vis_b,  kvis, nullptr, 192, 192, 0, 864,  192 * 864,  192 * 864,  0, 0, 0);
  GEMM(0, 0, vis_s,  nullptr, pv_vis_w,  pv_vis_b,  vvis, nullptr, 192, 192, 0, 864,  192 * 864,  192 * 864,  0, 0, 0);
  GEMM(0, 0, lwir_s, nullptr, pk_lwir_w, pk_lwir_b, klw,  nullptr, 192, 192, 0, 864,  192 * 864,  192 * 864,  0, 0, 0);
  GEMM(0, 0, lwir_s, nullptr, pv_lwir_w, pv_lwir_b, vlw,  nullptr, 192, 192, 0, 864,  192 * 864,  192 * 864,  0, 0, 0);
  // fused attention (12 heads)
  size_t shm = (2 * 16 * 872 + 32) * sizeof(float);
  attn_kernel<<<dim3(240, 12), 64, shm, stream>>>(qA, kvis, vvis, qinA);
  attn_kernel<<<dim3(240, 12), 64, shm, stream>>>(qB, klw,  vlw,  qinB);
  // output projections
  GEMM(0, 0, qinA, nullptr, pout_vis_w,  pout_vis_b,  qA, nullptr, 192, 192, 0, 7680, 192 * 7680, 192 * 7680, 0, 0, 0);
  GEMM(0, 0, qinB, nullptr, pout_lwir_w, pout_lwir_b, qB, nullptr, 192, 192, 0, 7680, 192 * 7680, 192 * 7680, 0, 0, 0);
  // final 192->512 + relu + residual, straight into d_out (stack order: vis, lwir)
  float* out = (float*)d_out;
  GEMM(0, 2, qA, nullptr, vout_w, vout_b, out,                    vis_x,
       512, 192, 0, 7680, 192 * 7680, 512 * 7680, 512 * 7680, 0, 0);
  GEMM(0, 2, qB, nullptr, lout_w, lout_b, out + 2ull * 512 * 7680, lwir_x,
       512, 192, 0, 7680, 192 * 7680, 512 * 7680, 512 * 7680, 0, 0);
  #undef GEMM
}
